// Fp16LinearAttention_67070209295097
// MI455X (gfx1250) — matrix operand order repaired and checked by reading, not compiled, not run on hardware
//
#include <hip/hip_runtime.h>
#include <math.h>

typedef __attribute__((ext_vector_type(16))) __bf16 v16bf;
typedef __attribute__((ext_vector_type(8)))  __bf16 v8bf;
typedef __attribute__((ext_vector_type(4)))  __bf16 v4bf;
typedef __attribute__((ext_vector_type(8)))  float  v8f;
typedef __attribute__((ext_vector_type(2)))  float  v2f;

#define BB 2
#define TT 2048
#define HH 16
#define DD 128
#define CC 64
#define NTHREADS 256
#define GLA_SCALE 0.08838834764831845f

// f32 staging strides (floats); bf16 tile strides (elements)
#define QS 132     // 528 B rows (async B128 target)
#define KS 132
#define VS 68      // 272 B rows (async B64 target)
#define QBS 136    // bf16, 272 B rows (16B-aligned k-runs)
#define KBS 136
#define TBS 72     // Vt bf16, 144 B rows
#define ABS 72     // Ab bf16, 144 B rows
#define SS 68      // state f32

// byte offsets into dynamic LDS (all 16B aligned)
#define OQ0 0
#define OQ1 (OQ0 + 64*QS*4)
#define OK0 (OQ1 + 64*QS*4)
#define OK1 (OK0 + 64*KS*4)
#define OV0 (OK1 + 64*KS*4)
#define OV1 (OV0 + 64*VS*4)
#define OQB (OV1 + 64*VS*4)
#define OKB (OQB + 64*QBS*2)
#define OVT (OKB + 64*KBS*2)
#define OAB (OVT + 64*TBS*2)
#define OS  (OAB + 64*ABS*2)
#define OG  (OS  + 128*SS*4)
#define SMEM_BYTES (OG + 72*4)

// ---- gfx1250 async global->LDS (ASYNCcnt-tracked) ----
__device__ __forceinline__ void async_b128(uint32_t lds_off, uint32_t goff, uint64_t saddr) {
  asm volatile("global_load_async_to_lds_b128 %0, %1, %2"
               :: "v"(lds_off), "v"(goff), "s"(saddr) : "memory");
}
__device__ __forceinline__ void async_b64(uint32_t lds_off, uint32_t goff, uint64_t saddr) {
  asm volatile("global_load_async_to_lds_b64 %0, %1, %2"
               :: "v"(lds_off), "v"(goff), "s"(saddr) : "memory");
}
__device__ __forceinline__ void wait_async0() {
  asm volatile("s_wait_asynccnt 0" ::: "memory");
}
__device__ __forceinline__ uint32_t lds_off32(const void* p) {
  return (uint32_t)(uintptr_t)p;
}

__device__ __forceinline__ void issue_chunk_loads(int t0, int wid, int lane, int vbase,
                                                  uint64_t qsa, uint64_t ksa, uint64_t vsa,
                                                  float* sQ, float* sK, float* sV) {
  const uint32_t qb = lds_off32(sQ), kb = lds_off32(sK), vb = lds_off32(sV);
#pragma unroll
  for (int i = 0; i < 8; ++i) {
    const int row = wid * 8 + i;
    const uint32_t go  = (uint32_t)(t0 + row) * (HH * DD * 4) + lane * 16;
    async_b128(qb + row * (QS * 4) + lane * 16, go, qsa);
    async_b128(kb + row * (KS * 4) + lane * 16, go, ksa);
    const uint32_t gov = (uint32_t)(t0 + row) * (HH * DD * 4) + vbase * 4 + lane * 8;
    async_b64(vb + row * (VS * 4) + lane * 8, gov, vsa);
  }
}

union B16x16 { v16bf v; v8bf h[2]; };

// bf16 A-operand (16x32): two contiguous 8-elem k-runs per lane
__device__ __forceinline__ v16bf ld_aop(const __bf16* rowbase, int hi) {
  B16x16 u;
  u.h[0] = *(const v8bf*)(rowbase + hi * 8);
  u.h[1] = *(const v8bf*)(rowbase + 16 + hi * 8);
  return u.v;
}
// bf16 B-operand (32x16): two contiguous 8-elem k-runs per lane
__device__ __forceinline__ v16bf ld_bop(const __bf16* rowbase, int hi) {
  B16x16 u;
  u.h[0] = *(const v8bf*)(rowbase + hi * 8);
  u.h[1] = *(const v8bf*)(rowbase + 16 + hi * 8);
  return u.v;
}

__global__ void gla_chunk_kernel(const float* __restrict__ q,
                                 const float* __restrict__ k,
                                 const float* __restrict__ v,
                                 const float* __restrict__ s0,
                                 float* __restrict__ out) {
  extern __shared__ char smemb[];
  __bf16* Qb  = (__bf16*)(smemb + OQB);
  __bf16* Kb  = (__bf16*)(smemb + OKB);
  __bf16* Vt  = (__bf16*)(smemb + OVT);   // transposed: Vt[dv][t]
  __bf16* Ab  = (__bf16*)(smemb + OAB);
  float*  sS  = (float*)(smemb + OS);     // persistent state slice (f32)
  float*  gpow = (float*)(smemb + OG);

  const int tid  = threadIdx.x;
  const int lane = tid & 31;
  const int wid  = tid >> 5;
  const int hi   = lane >> 4;
  const int lm   = lane & 15;

  const int blk   = blockIdx.x;
  const int vh    = blk & 1;
  const int bh    = blk >> 1;
  const int h     = bh & (HH - 1);
  const int b     = bh >> 4;
  const int vbase = vh * 64;

  const uint64_t qsa = (uint64_t)(uintptr_t)(q + ((size_t)b * TT * HH + h) * DD);
  const uint64_t ksa = (uint64_t)(uintptr_t)(k + ((size_t)b * TT * HH + h) * DD);
  const uint64_t vsa = (uint64_t)(uintptr_t)(v + ((size_t)b * TT * HH + h) * DD);

  issue_chunk_loads(0, wid, lane, vbase, qsa, ksa, vsa,
                    (float*)(smemb + OQ0), (float*)(smemb + OK0), (float*)(smemb + OV0));

  const float slope = exp2f(-0.5f * (float)(h + 1));
  const float sdec  = -slope * (1.0f - 12.0f / 31.0f + 1e-5f);
  const float gdec  = __expf(sdec);            // gamma, kept in a register
  if (tid < 65) gpow[tid] = __expf(sdec * (float)tid);

  {
    const float* sp = s0 + ((size_t)(b * HH + h) * DD) * DD + vbase;
    for (int idx = tid; idx < 128 * 16; idx += NTHREADS) {
      int r = idx >> 4, c4 = idx & 15;
      *(float4*)(sS + r * SS + c4 * 4) = *(const float4*)(sp + (size_t)r * DD + c4 * 4);
    }
  }
  wait_async0();
  __syncthreads();

  int p = 0;
  for (int c = 0; c < TT / CC; ++c) {
    const int t0 = c * CC;
    const float* sQf = (const float*)(smemb + (p ? OQ1 : OQ0));
    const float* sKf = (const float*)(smemb + (p ? OK1 : OK0));
    const float* sVf = (const float*)(smemb + (p ? OV1 : OV0));

    // prefetch next chunk into the other staging buffer (overlaps everything below)
    if (c + 1 < TT / CC)
      issue_chunk_loads(t0 + CC, wid, lane, vbase, qsa, ksa, vsa,
                        (float*)(smemb + (p ? OQ0 : OQ1)),
                        (float*)(smemb + (p ? OK0 : OK1)),
                        (float*)(smemb + (p ? OV0 : OV1)));

    // ---- pack staged f32 -> bf16 tiles (Q,K row-major; V transposed) ----
    for (int idx = tid; idx < 64 * 32; idx += NTHREADS) {
      int r = idx >> 5, c4 = (idx & 31) * 4;
      float4 fq = *(const float4*)(sQf + r * QS + c4);
      float4 fk = *(const float4*)(sKf + r * KS + c4);
      v4bf bq, bk;
      bq[0] = (__bf16)fq.x; bq[1] = (__bf16)fq.y; bq[2] = (__bf16)fq.z; bq[3] = (__bf16)fq.w;
      bk[0] = (__bf16)fk.x; bk[1] = (__bf16)fk.y; bk[2] = (__bf16)fk.z; bk[3] = (__bf16)fk.w;
      *(v4bf*)(Qb + r * QBS + c4) = bq;
      *(v4bf*)(Kb + r * KBS + c4) = bk;
    }
    for (int idx = tid; idx < 64 * 16; idx += NTHREADS) {
      int dv = idx >> 4, tg = (idx & 15) * 4;
      v4bf bt;
      bt[0] = (__bf16)sVf[(tg + 0) * VS + dv];
      bt[1] = (__bf16)sVf[(tg + 1) * VS + dv];
      bt[2] = (__bf16)sVf[(tg + 2) * VS + dv];
      bt[3] = (__bf16)sVf[(tg + 3) * VS + dv];
      *(v4bf*)(Vt + dv * TBS + tg) = bt;
    }
    __syncthreads();

    // ---- A = Q @ K^T (bf16 WMMA), causal decay mask -> Ab (bf16) ----
#pragma unroll
    for (int u = 0; u < 2; ++u) {
      int tile = wid * 2 + u;
      int tm = tile >> 2, tn = tile & 3;
      v8f acc = {};
#pragma unroll
      for (int kk = 0; kk < 4; ++kk) {
        v16bf av = ld_aop(Qb + (tm * 16 + lm) * QBS + kk * 32, hi);
        v16bf bv = ld_bop(Kb + (tn * 16 + lm) * KBS + kk * 32, hi);
        acc = __builtin_amdgcn_wmma_f32_16x16x32_bf16(false, av, false, bv,
                                                      (short)0, acc, false, false);
      }
      // w(r) = SCALE * gamma^(d0+r), geometric in r: 1 expf + 1 mul/row, no LDS
      const int d0 = (tm - tn) * 16 + 8 * hi - lm;   // t - i at r = 0
      float w = __expf(sdec * (float)d0) * GLA_SCALE;
#pragma unroll
      for (int r = 0; r < 8; ++r) {
        int t = tm * 16 + r + 8 * hi;
        float val = acc[r] * w;
        val = (d0 + r >= 0) ? val : 0.0f;            // v_cndmask, no exec branches
        Ab[t * ABS + tn * 16 + lm] = (__bf16)val;
        w *= gdec;
      }
    }

    // ---- O_inter = Q @ S_prev (f32 WMMA), rows scaled by SCALE*gamma^{t+1} ----
    v8f oacc[2];
#pragma unroll
    for (int u = 0; u < 2; ++u) {
      int tile = wid * 2 + u;
      int tm = tile >> 2, tn = tile & 3;
      v8f acc = {};
      for (int kk = 0; kk < 32; ++kk) {
        int kb = kk * 4 + hi * 2;
        v2f av, bv;
        av[0] = sQf[(tm * 16 + lm) * QS + kb];
        av[1] = sQf[(tm * 16 + lm) * QS + kb + 1];
        bv[0] = sS[kb * SS + tn * 16 + lm];
        bv[1] = sS[(kb + 1) * SS + tn * 16 + lm];
        acc = __builtin_amdgcn_wmma_f32_16x16x4_f32(false, av, false, bv,
                                                    (short)0, acc, false, false);
      }
      float w = __expf(sdec * (float)(tm * 16 + 8 * hi + 1)) * GLA_SCALE;
#pragma unroll
      for (int r = 0; r < 8; ++r) {
        acc[r] *= w;
        w *= gdec;
      }
      oacc[u] = acc;
    }
    __syncthreads();  // Ab ready; all S_prev reads done

    // ---- O += A~ @ V (bf16 WMMA via Ab and transposed Vt), store ----
#pragma unroll
    for (int u = 0; u < 2; ++u) {
      int tile = wid * 2 + u;
      int tm = tile >> 2, tn = tile & 3;
      v8f acc = oacc[u];
#pragma unroll
      for (int kk = 0; kk < 2; ++kk) {
        v16bf av = ld_aop(Ab + (tm * 16 + lm) * ABS + kk * 32, hi);
        v16bf bv = ld_bop(Vt + (tn * 16 + lm) * TBS + kk * 32, hi);
        acc = __builtin_amdgcn_wmma_f32_16x16x32_bf16(false, av, false, bv,
                                                      (short)0, acc, false, false);
      }
#pragma unroll
      for (int r = 0; r < 8; ++r) {
        int t = tm * 16 + r + 8 * hi;
        size_t g = (((size_t)b * TT + t0 + t) * HH + h) * DD + vbase + tn * 16 + lm;
        out[g] = acc[r];
      }
    }

    // ---- S = gamma^64 * S + (K .* gamma^{63-t})^T @ V  (f32 WMMA) ----
    const float gC = gpow[64];
#pragma unroll
    for (int u = 0; u < 4; ++u) {
      int tile = wid * 4 + u;
      int tm = tile >> 2, tn = tile & 3;
      v8f acc;
#pragma unroll
      for (int r = 0; r < 8; ++r)
        acc[r] = sS[(tm * 16 + r + 8 * hi) * SS + tn * 16 + lm] * gC;
      for (int kk = 0; kk < 16; ++kk) {
        int kb = kk * 4 + hi * 2;
        v2f av, bv;
        av[0] = sKf[kb * KS + tm * 16 + lm] * gpow[63 - kb];
        av[1] = sKf[(kb + 1) * KS + tm * 16 + lm] * gpow[62 - kb];
        bv[0] = sVf[kb * VS + tn * 16 + lm];
        bv[1] = sVf[(kb + 1) * VS + tn * 16 + lm];
        acc = __builtin_amdgcn_wmma_f32_16x16x4_f32(false, av, false, bv,
                                                    (short)0, acc, false, false);
      }
#pragma unroll
      for (int r = 0; r < 8; ++r)
        sS[(tm * 16 + r + 8 * hi) * SS + tn * 16 + lm] = acc[r];
    }

    wait_async0();   // next chunk staged
    __syncthreads();
    p ^= 1;
  }
}

extern "C" void kernel_launch(void* const* d_in, const int* in_sizes, int n_in,
                              void* d_out, int out_size, void* d_ws, size_t ws_size,
                              hipStream_t stream) {
  (void)in_sizes; (void)n_in; (void)d_ws; (void)ws_size; (void)out_size;
  const float* q  = (const float*)d_in[0];
  const float* k  = (const float*)d_in[1];
  const float* v  = (const float*)d_in[2];
  const float* s0 = (const float*)d_in[3];
  float* out = (float*)d_out;

  const size_t shmem = SMEM_BYTES;  // ~258 KB < 320 KB/WGP
  (void)hipFuncSetAttribute((const void*)gla_chunk_kernel,
                            hipFuncAttributeMaxDynamicSharedMemorySize, (int)shmem);

  dim3 grid(BB * HH * 2);   // (b, h, Dv-half) -> 64 workgroups
  dim3 block(NTHREADS);     // 8 waves (wave32)
  gla_chunk_kernel<<<grid, block, shmem, stream>>>(q, k, v, s0, out);
}
